// GATLayer_28467043238495
// MI455X (gfx1250) — matrix operand order repaired and checked by reading, not compiled, not run on hardware
//
#include <hip/hip_runtime.h>

#define BB 8
#define NN 1024
#define IND 512
#define HH 8
#define SUPD 64   // per-head output dim
#define ODIM (HH*SUPD)  // 512

typedef __attribute__((ext_vector_type(16))) __bf16 v16bf;
typedef __attribute__((ext_vector_type(8)))  float  v8f;

union BfVec {
  v16bf v;
  unsigned short h[16];
  uint4 q[2];
};

__device__ __forceinline__ unsigned short f2bf(float f) {
  unsigned int u = __float_as_uint(f);
  u += 0x7fffu + ((u >> 16) & 1u);   // round-to-nearest-even
  return (unsigned short)(u >> 16);
}

__device__ __forceinline__ float lrelu(float x) { return x > 0.f ? x : 0.01f * x; }

// async DMA: global -> LDS, 16 bytes per lane, tracked with ASYNCcnt (CDNA5)
__device__ __forceinline__ void async_g2l_b128(unsigned lds_byte_addr,
                                               const void* gaddr) {
  unsigned long long ga = (unsigned long long)(uintptr_t)gaddr;
  asm volatile("global_load_async_to_lds_b128 %0, %1, off"
               :: "v"(lds_byte_addr), "v"(ga) : "memory");
}
__device__ __forceinline__ void wait_asynccnt0() {
  asm volatile("s_wait_asynccnt 0x0" ::: "memory");
}

// ---------------- convert X (f32 -> bf16), elementwise ----------------
__global__ void __launch_bounds__(256) cvt_x_kernel(const float* __restrict__ X,
                                                   unsigned short* __restrict__ Xb) {
  int i = (blockIdx.x * 256 + threadIdx.x) * 4;
  float4 f = *(const float4*)(X + i);
  unsigned long long p = (unsigned long long)f2bf(f.x)
                       | ((unsigned long long)f2bf(f.y) << 16)
                       | ((unsigned long long)f2bf(f.z) << 32)
                       | ((unsigned long long)f2bf(f.w) << 48);
  *(unsigned long long*)(Xb + i) = p;
}

// ------------- convert + transpose W: [H,IN,SUP] -> Wt[H,SUP,IN] bf16 -------------
__global__ void __launch_bounds__(256) cvt_w_kernel(const float* __restrict__ W,
                                                    unsigned short* __restrict__ Wt) {
  int t = blockIdx.x * 256 + threadIdx.x;       // linear index in Wt
  int i = t % IND;
  int o = (t / IND) % SUPD;
  int h = t / (IND * SUPD);
  Wt[t] = f2bf(W[((size_t)h * IND + i) * SUPD + o]);
}

// ---------------- projection: inputs = X@W (bf16 WMMA), plus self/neigh logits ----------------
__global__ void __launch_bounds__(256) proj_kernel(
    const unsigned short* __restrict__ Xb,   // [B,N,IN] bf16
    const unsigned short* __restrict__ Wt,   // [H,SUP,IN] bf16
    const float* __restrict__ w1, const float* __restrict__ b1,
    const float* __restrict__ w2, const float* __restrict__ b2,
    unsigned short* __restrict__ Vt,         // [B,H,SUP,N] bf16 (transposed inputs)
    float* __restrict__ selfF,               // [B,H,N]
    float* __restrict__ neighF) {            // [B,H,N]
  const int lane = threadIdx.x & 31;
  const int wid  = blockIdx.x * 8 + (threadIdx.x >> 5);   // 4096 waves
  const int nt = wid & 63;
  const int h  = (wid >> 6) & 7;
  const int b  = wid >> 9;
  const int row0 = nt * 16;
  const int r  = lane & 15;
  const int hi = lane >> 4;

  v8f acc[4];
#pragma unroll
  for (int ct = 0; ct < 4; ++ct)
#pragma unroll
    for (int j = 0; j < 8; ++j) acc[ct][j] = 0.f;

  const unsigned short* xrow = Xb + (size_t)(b * NN + row0 + r) * IND;
#pragma unroll 2
  for (int k0 = 0; k0 < IND; k0 += 32) {
    BfVec a;
    // 16-bit A-operand layout: lanes<16 hold K=k0+{0..7,16..23}, lanes>=16 hold +8
    a.q[0] = *(const uint4*)(xrow + k0 + 8 * hi);
    a.q[1] = *(const uint4*)(xrow + k0 + 16 + 8 * hi);
#pragma unroll
    for (int ct = 0; ct < 4; ++ct) {
      BfVec bm;
      const unsigned short* wrow =
          Wt + (size_t)(h * SUPD + ct * 16 + r) * IND + k0 + 8 * hi;
      bm.q[0] = *(const uint4*)(wrow);
      bm.q[1] = *(const uint4*)(wrow + 16);
      acc[ct] = __builtin_amdgcn_wmma_f32_16x16x32_bf16(
          false, a.v, false, bm.v, (short)0, acc[ct], false, false);
    }
  }

  // self_fc / neigh_fc: per-row dot with w1/w2 (cross-lane reduce over 16 cols)
  float w1v[4], w2v[4];
#pragma unroll
  for (int ct = 0; ct < 4; ++ct) {
    w1v[ct] = w1[h * SUPD + ct * 16 + r];
    w2v[ct] = w2[h * SUPD + ct * 16 + r];
  }
  float p1[8], p2[8];
#pragma unroll
  for (int j = 0; j < 8; ++j) {
    float s1 = 0.f, s2 = 0.f;
#pragma unroll
    for (int ct = 0; ct < 4; ++ct) {
      s1 += acc[ct][j] * w1v[ct];
      s2 += acc[ct][j] * w2v[ct];
    }
    p1[j] = s1; p2[j] = s2;
  }
#pragma unroll
  for (int m = 1; m < 16; m <<= 1) {
#pragma unroll
    for (int j = 0; j < 8; ++j) {
      p1[j] += __shfl_xor(p1[j], m, 32);
      p2[j] += __shfl_xor(p2[j], m, 32);
    }
  }
  if (r == 0) {  // lanes 0 (rows 0..7) and 16 (rows 8..15)
    float bb1 = b1[h], bb2 = b2[h];
    size_t base = (size_t)(b * HH + h) * NN + row0 + 8 * hi;
#pragma unroll
    for (int j = 0; j < 8; ++j) {
      selfF[base + j]  = p1[j] + bb1;
      neighF[base + j] = p2[j] + bb2;
    }
  }

  // store transposed V (bf16): Vt[b,h,o, n] ; C/D layout => 8 consecutive n per lane
#pragma unroll
  for (int ct = 0; ct < 4; ++ct) {
    uint4 st;
    st.x = (unsigned)f2bf(acc[ct][0]) | ((unsigned)f2bf(acc[ct][1]) << 16);
    st.y = (unsigned)f2bf(acc[ct][2]) | ((unsigned)f2bf(acc[ct][3]) << 16);
    st.z = (unsigned)f2bf(acc[ct][4]) | ((unsigned)f2bf(acc[ct][5]) << 16);
    st.w = (unsigned)f2bf(acc[ct][6]) | ((unsigned)f2bf(acc[ct][7]) << 16);
    *(uint4*)(Vt + (size_t)((b * HH + h) * SUPD + ct * 16 + r) * NN + row0 + 8 * hi) = st;
  }
}

// ---------------- attention: softmax(logits+mask) @ V, A shared across heads via LDS ----------------
__global__ void __launch_bounds__(256) attn_kernel(
    const float* __restrict__ A,             // [B,N,N] 0/1
    const unsigned short* __restrict__ Vt,   // [B,H,SUP,N] bf16
    const float* __restrict__ selfF,         // [B,H,N]
    const float* __restrict__ neighF,        // [B,H,N]
    float* __restrict__ out) {               // [B,N,H*SUP]
  __shared__ __align__(16) float neighS[HH * NN];        // 32 KB (async DMA filled)
  __shared__ __align__(16) unsigned char Atile[16 * NN]; // 16 KB (f32->u8 compressed)
  __shared__ float selfS[HH * 16];

  const int b  = blockIdx.x >> 6;
  const int nt = blockIdx.x & 63;
  const int row0 = nt * 16;
  const int t = threadIdx.x;
  const int wid = t >> 5, lane = t & 31;
  const int r = lane & 15, hi = lane >> 4;

  // stage neigh logits for all 8 heads via async global->LDS DMA (ASYNCcnt path)
  {
    const float* src = neighF + (size_t)(b * HH + wid) * NN;
    unsigned dbase = (unsigned)(uintptr_t)(&neighS[wid * NN]);  // addr[31:0] == LDS offset
#pragma unroll
    for (int i = lane * 4; i < NN; i += 128) {
      async_g2l_b128(dbase + i * 4, src + i);
    }
  }
  // stage A tile as u8 (read A from HBM exactly once per (b,row-tile))
  {
    int row = t >> 4;
    int cbase = (t & 15) * 64;
    const float* arow = A + ((size_t)b * NN + row0 + row) * NN;
#pragma unroll
    for (int i = 0; i < 64; i += 4) {
      float4 f = *(const float4*)(arow + cbase + i);
      uchar4 u;
      u.x = (unsigned char)(f.x != 0.f);
      u.y = (unsigned char)(f.y != 0.f);
      u.z = (unsigned char)(f.z != 0.f);
      u.w = (unsigned char)(f.w != 0.f);
      *(uchar4*)&Atile[row * NN + cbase + i] = u;
    }
  }
  if (t < HH * 16)
    selfS[t] = selfF[(size_t)(b * HH + (t >> 4)) * NN + row0 + (t & 15)];
  wait_asynccnt0();     // async LDS writes visible before barrier
  __syncthreads();

  const int h = wid;
  const float sfr = selfS[h * 16 + r];
  const unsigned short* vbase = Vt + (size_t)(b * HH + h) * SUPD * NN;
#pragma unroll
  for (int ct = 0; ct < 4; ++ct)
    __builtin_prefetch(vbase + (size_t)(ct * 16 + r) * NN, 0, 1);

  // pass 1: row max of masked leaky-relu logits
  float mx = -3.0e38f;
  {
    const int c0 = hi * 512;
    for (int c = c0; c < c0 + 512; c += 4) {
      uchar4 a = *(const uchar4*)&Atile[r * NN + c];
      float4 n = *(const float4*)&neighS[h * NN + c];
      float l0 = lrelu(sfr + n.x) + (a.x ? 0.f : -1e9f);
      float l1 = lrelu(sfr + n.y) + (a.y ? 0.f : -1e9f);
      float l2 = lrelu(sfr + n.z) + (a.z ? 0.f : -1e9f);
      float l3 = lrelu(sfr + n.w) + (a.w ? 0.f : -1e9f);
      mx = fmaxf(mx, fmaxf(fmaxf(l0, l1), fmaxf(l2, l3)));
    }
    mx = fmaxf(mx, __shfl_xor(mx, 16, 32));
  }

  // pass 2: P = exp(logit - mx), accumulate P@V with bf16 WMMA
  v8f O[4];
#pragma unroll
  for (int ct = 0; ct < 4; ++ct)
#pragma unroll
    for (int j = 0; j < 8; ++j) O[ct][j] = 0.f;
  float rowsum = 0.f;

  for (int k0 = 0; k0 < NN; k0 += 32) {
    BfVec P;
    const int cA = k0 + 8 * hi;
    const int cB = k0 + 16 + 8 * hi;
    uchar4 aA0 = *(const uchar4*)&Atile[r * NN + cA];
    uchar4 aA1 = *(const uchar4*)&Atile[r * NN + cA + 4];
    uchar4 aB0 = *(const uchar4*)&Atile[r * NN + cB];
    uchar4 aB1 = *(const uchar4*)&Atile[r * NN + cB + 4];
    float4 nA0 = *(const float4*)&neighS[h * NN + cA];
    float4 nA1 = *(const float4*)&neighS[h * NN + cA + 4];
    float4 nB0 = *(const float4*)&neighS[h * NN + cB];
    float4 nB1 = *(const float4*)&neighS[h * NN + cB + 4];

#define MKP(idx, av, nv)                                      \
    {                                                         \
      float l = lrelu(sfr + (nv));                            \
      l = (av) ? l : l - 1e9f;                                \
      float p = __expf(l - mx);                               \
      rowsum += p;                                            \
      P.h[idx] = f2bf(p);                                     \
    }
    MKP(0, aA0.x, nA0.x) MKP(1, aA0.y, nA0.y) MKP(2, aA0.z, nA0.z) MKP(3, aA0.w, nA0.w)
    MKP(4, aA1.x, nA1.x) MKP(5, aA1.y, nA1.y) MKP(6, aA1.z, nA1.z) MKP(7, aA1.w, nA1.w)
    MKP(8, aB0.x, nB0.x) MKP(9, aB0.y, nB0.y) MKP(10, aB0.z, nB0.z) MKP(11, aB0.w, nB0.w)
    MKP(12, aB1.x, nB1.x) MKP(13, aB1.y, nB1.y) MKP(14, aB1.z, nB1.z) MKP(15, aB1.w, nB1.w)
#undef MKP

#pragma unroll
    for (int ct = 0; ct < 4; ++ct) {
      const unsigned short* vr =
          vbase + (size_t)(ct * 16 + r) * NN + k0 + 8 * hi;
      BfVec Vb;
      Vb.q[0] = *(const uint4*)vr;
      Vb.q[1] = *(const uint4*)(vr + 16);
      O[ct] = __builtin_amdgcn_wmma_f32_16x16x32_bf16(
          false, P.v, false, Vb.v, (short)0, O[ct], false, false);
    }
  }

  rowsum += __shfl_xor(rowsum, 16, 32);
  float inv = 1.0f / rowsum;
  float invs[8];
#pragma unroll
  for (int j = 0; j < 8; ++j) invs[j] = __shfl(inv, j + 8 * hi, 32);

  // store: out[b, n, h*64 + o], relu applied
#pragma unroll
  for (int ct = 0; ct < 4; ++ct) {
#pragma unroll
    for (int j = 0; j < 8; ++j) {
      float v = fmaxf(O[ct][j] * invs[j], 0.f);
      out[(size_t)(b * NN + row0 + j + 8 * hi) * ODIM + h * SUPD + ct * 16 + r] = v;
    }
  }
}

extern "C" void kernel_launch(void* const* d_in, const int* in_sizes, int n_in,
                              void* d_out, int out_size, void* d_ws, size_t ws_size,
                              hipStream_t stream) {
  const float* A  = (const float*)d_in[0];
  const float* X  = (const float*)d_in[1];
  const float* W  = (const float*)d_in[2];
  const float* w1 = (const float*)d_in[3];
  const float* b1 = (const float*)d_in[4];
  const float* w2 = (const float*)d_in[5];
  const float* b2 = (const float*)d_in[6];
  float* out = (float*)d_out;

  char* ws = (char*)d_ws;
  unsigned short* Xb = (unsigned short*)ws;                       // 8 MB
  unsigned short* Wt = (unsigned short*)(ws + 8388608);           // 512 KB
  unsigned short* Vt = (unsigned short*)(ws + 8388608 + 524288);  // 8 MB
  float* selfF  = (float*)(ws + 8388608 + 524288 + 8388608);      // 256 KB
  float* neighF = (float*)(ws + 8388608 + 524288 + 8388608 + 262144);

  cvt_x_kernel<<<(BB * NN * IND) / (256 * 4), 256, 0, stream>>>(X, Xb);
  cvt_w_kernel<<<(HH * IND * SUPD) / 256, 256, 0, stream>>>(W, Wt);
  proj_kernel<<<BB * HH * (NN / 16) / 8, 256, 0, stream>>>(
      Xb, Wt, w1, b1, w2, b2, Vt, selfF, neighF);
  attn_kernel<<<BB * (NN / 16), 256, 0, stream>>>(A, Vt, selfF, neighF, out);
}
